// FeatureExtractor_33002528703090
// MI455X (gfx1250) — matrix operand order repaired and checked by reading, not compiled, not run on hardware
//
#include <hip/hip_runtime.h>
#include <hip/hip_bf16.h>
#include <hip/hip_fp16.h>

// ---------------------------------------------------------------------------
// MI455X / gfx1250 implementation of the DGCNN-style feature extractor.
// All matrix contractions (KNN Gram matrices, edge-MLP layers, prep convs)
// are lowered to v_wmma_f32_16x16x32_f16 with f32 accumulation. Wave32.
// Candidate tiles for KNN are staged memory->LDS with the CDNA5 async
// engine (GLOBAL_LOAD_ASYNC_TO_LDS_B128 + s_wait_asynccnt).
// ---------------------------------------------------------------------------

typedef __attribute__((ext_vector_type(16))) _Float16 v16h;
typedef __attribute__((ext_vector_type(8)))  float    v8f;

#define BB 4
#define NN 4096
#define KNN_K 16

static __device__ inline v8f vzero8() {
  v8f v = {0.f, 0.f, 0.f, 0.f, 0.f, 0.f, 0.f, 0.f};
  return v;
}

static __device__ inline v8f wmma_f16(v16h a, v16h b, v8f c) {
  // D = A(16x32,f16) * B(32x16,f16) + C(16x16,f32)
  return __builtin_amdgcn_wmma_f32_16x16x32_f16(
      /*neg_a=*/false, a, /*neg_b=*/false, b,
      /*c_mod=*/(short)0, c, /*reuse_a=*/false, /*reuse_b=*/false);
}

// LDS byte offset of a __shared__ object: flat->LDS aperture mapping uses
// addr[31:0], so truncation of the generic address yields the DS address.
static __device__ inline unsigned lds_off(const void* p) {
  return (unsigned)(unsigned long long)(uintptr_t)p;
}

// Async memory->LDS copy of 16 bytes (CDNA5, ASYNCcnt-tracked).
static __device__ inline void async_copy_b128(unsigned ldsaddr, const void* g) {
  unsigned long long ga = (unsigned long long)(uintptr_t)g;
  asm volatile("global_load_async_to_lds_b128 %0, %1, off"
               :: "v"(ldsaddr), "v"(ga)
               : "memory");
}

static __device__ inline void wait_async0() {
  asm volatile("s_wait_asynccnt 0x0" ::: "memory");
}

// Build an A fragment (16x32 f16) from an LDS matrix of 16 rows x ldy cols,
// with a 2-segment logical-K -> physical-column map:
//   logical k in [0,L0)      -> column o0 + k
//   logical k in [L0,L0+L1)  -> column o1 + (k - L0)
//   else                     -> 0
// A k order used here (contiguous, same as B): lanes 0-15 hold M=lane, K in {0..15}+kbase;
// lanes 16-31 hold M=lane-16, K in {16..31}+kbase.
static __device__ inline v16h load_a_frag(const _Float16* y, int ldy, int kbase,
                                          int o0, int L0, int o1, int L1,
                                          int lane) {
  v16h a;
  const int m  = lane & 15;
  const int hi = (lane >> 4) & 1;
  const _Float16* row = y + m * ldy;
#pragma unroll
  for (int e = 0; e < 16; ++e) {
    const int kk = hi * 16 + e;
    const int k  = kbase + kk;
    _Float16 v = (_Float16)0.0f;
    if (k < L0)            v = row[o0 + k];
    else if (k < L0 + L1)  v = row[o1 + (k - L0)];
    a[e] = v;
  }
  return a;
}

// B fragment (32x16 f16) from a row-major f32 weight matrix w[Kdim][Nw].
// B k order used here (contiguous, same as A): lane holds column n=lane&15; lanes 0-15 carry
// K=kbase+0..15, lanes 16-31 carry K=kbase+16..31.
static __device__ inline v16h load_b_frag_w(const float* w, int Kdim, int Nw,
                                            int kbase, int nbase, int lane) {
  v16h b;
  const int n  = nbase + (lane & 15);
  const int hi = (lane >> 4) & 1;
#pragma unroll
  for (int e = 0; e < 16; ++e) {
    const int k = kbase + hi * 16 + e;
    float v = 0.f;
    if (k < Kdim && n < Nw) v = w[k * Nw + n];
    b[e] = (_Float16)v;
  }
  return b;
}

// B fragment = transpose of an LDS candidate tile (16 rows x ldc cols, f32,
// zero padded): B[k][n] = (f16)tile[n][k].
static __device__ inline v16h load_b_frag_lds_T_f32(const float* c, int ldc,
                                                    int kbase, int lane) {
  v16h b;
  const int n  = lane & 15;
  const int hi = (lane >> 4) & 1;
  const float* row = c + n * ldc;
#pragma unroll
  for (int e = 0; e < 16; ++e) b[e] = (_Float16)row[kbase + hi * 16 + e];
  return b;
}

// ---------------------------------------------------------------------------
// layer0: (B,3,N) -> Xall[:, :, 456:480] = x @ l0_w + l0_b   (linear, 3->24)
// ---------------------------------------------------------------------------
__global__ void layer0_kernel(const float* __restrict__ xyz,
                              const float* __restrict__ w,
                              const float* __restrict__ bias,
                              float* __restrict__ Xall) {
  const int p = blockIdx.x * blockDim.x + threadIdx.x;
  if (p >= BB * NN) return;
  const int b = p / NN, n = p % NN;
  const float x0 = xyz[((size_t)b * 3 + 0) * NN + n];
  const float x1 = xyz[((size_t)b * 3 + 1) * NN + n];
  const float x2 = xyz[((size_t)b * 3 + 2) * NN + n];
  float* dst = Xall + (size_t)p * 480 + 456;
#pragma unroll
  for (int c = 0; c < 24; ++c)
    dst[c] = bias[c] + x0 * w[c] + x1 * w[24 + c] + x2 * w[48 + c];
}

// ---------------------------------------------------------------------------
// KNN: one wave per 16-query tile. Gram tile via WMMA, top-16 per query.
// feat: (B*N, fstride) f32, use first C columns. idx_out: (B*N,16) int.
// Candidate tiles staged with GLOBAL_LOAD_ASYNC_TO_LDS_B128.
// Requires C % 4 == 0 (true: 24 or 48).
// ---------------------------------------------------------------------------
__global__ __launch_bounds__(32) void knn_kernel(
    const float* __restrict__ feat, int fstride, int C,
    int* __restrict__ idx_out) {
  __shared__ _Float16 sQ[16][64];
  __shared__ float sCf[16][64];
  __shared__ float sQsq[16];
  __shared__ float sCsq[16];
  __shared__ float sD[16][16];

  const int lane = threadIdx.x;
  const int tilesPerBatch = NN / 16;
  const int b  = blockIdx.x / tilesPerBatch;
  const int q0 = (blockIdx.x % tilesPerBatch) * 16;
  const float* fb = feat + (size_t)b * NN * fstride;

  // zero-pad both staging tiles once (pad columns stay zero)
  for (int t = lane; t < 16 * 64; t += 32) {
    (&sQ[0][0])[t]  = (_Float16)0.f;
    (&sCf[0][0])[t] = 0.f;
  }
  __syncthreads();
  for (int t = lane; t < 16 * C; t += 32) {
    const int r = t / C, c = t % C;
    sQ[r][c] = (_Float16)fb[(size_t)(q0 + r) * fstride + c];
  }
  __syncthreads();
  if (lane < 16) {
    float s = 0.f;
    for (int c = 0; c < C; ++c) { const float v = (float)sQ[lane][c]; s += v * v; }
    sQsq[lane] = s;
  }
  __syncthreads();

  const int nks = (C > 32) ? 2 : 1;  // block-uniform
  const v16h aF0 = load_a_frag(&sQ[0][0], 64, 0,  0, C, 0, 0, lane);
  const v16h aF1 = load_a_frag(&sQ[0][0], 64, 32, 0, C, 0, 0, lane);

  const int chunksPerRow = C >> 2;          // 16-byte chunks per row
  const int totChunks    = 16 * chunksPerRow;

  float bestD[KNN_K];
  int   bestI[KNN_K];
#pragma unroll
  for (int j = 0; j < KNN_K; ++j) { bestD[j] = 3.0e38f; bestI[j] = 0; }

  for (int ct = 0; ct < tilesPerBatch; ++ct) {
    const int c0 = ct * 16;
    if (ct + 1 < tilesPerBatch)  // gfx1250 global_prefetch_b8 for next tile
      __builtin_prefetch(fb + (size_t)(c0 + 16) * fstride, 0, 0);
    __syncthreads();
    // async memory->LDS staging of the candidate tile (f32, ASYNCcnt)
    for (int q = lane; q < totChunks; q += 32) {
      const int r = q / chunksPerRow, c4 = q % chunksPerRow;
      async_copy_b128(lds_off(&sCf[r][c4 * 4]),
                      fb + (size_t)(c0 + r) * fstride + c4 * 4);
    }
    wait_async0();
    __syncthreads();
    if (lane < 16) {
      float s = 0.f;
      for (int c = 0; c < C; ++c) { const float v = sCf[lane][c]; s += v * v; }
      sCsq[lane] = s;
    }
    __syncthreads();

    v8f acc = vzero8();
    acc = wmma_f16(aF0, load_b_frag_lds_T_f32(&sCf[0][0], 64, 0, lane), acc);
    if (nks == 2)
      acc = wmma_f16(aF1, load_b_frag_lds_T_f32(&sCf[0][0], 64, 32, lane), acc);

    const int n  = lane & 15;
    const int mb = (lane >> 4) * 8;
#pragma unroll
    for (int r = 0; r < 8; ++r) {
      const int m = mb + r;
      sD[m][n] = sQsq[m] + sCsq[n] - 2.0f * acc[r];
    }
    __syncthreads();
    if (lane < 16) {
      const int m = lane;
#pragma clang loop unroll(disable)
      for (int n2 = 0; n2 < 16; ++n2) {
        const int cand = c0 + n2;
        if (cand == q0 + m) continue;  // exclude self
        const float d = sD[m][n2];
        if (d < bestD[KNN_K - 1]) {
          int j = KNN_K - 1;
#pragma clang loop unroll(disable)
          while (j > 0 && bestD[j - 1] > d) {
            bestD[j] = bestD[j - 1]; bestI[j] = bestI[j - 1]; --j;
          }
          bestD[j] = d; bestI[j] = cand;
        }
      }
    }
  }
  __syncthreads();
  if (lane < 16) {
    int* dst = idx_out + ((size_t)(b * NN + q0 + lane)) * KNN_K;
#pragma unroll
    for (int j = 0; j < KNN_K; ++j) dst[j] = bestI[j];
  }
}

// ---------------------------------------------------------------------------
// DenseEdgeConv: one wave per point (its 16 neighbors = one WMMA M-tile).
// 3 MLP layers with zero-copy concat via 2-segment column maps, then
// max over 16 neighbors. Output Cout = 72 + C written at outbuf (stride 480).
// LDS slab column plan (width 160):
//   y0:  ctr -> [0,C),  nbr-ctr -> [C,2C)
//   h0 (relu) -> [2C+24, 2C+48)
//   h1 (relu) -> [2C,    2C+24)
//   h2 (lin)  -> [2C-24, 2C)      (overwrites tail of nbr-ctr, no longer used)
//   final y  = [[2C-24, 2C+48) (=72 cols: h2|h1|h0], [0,C) (=ctr)]
// ---------------------------------------------------------------------------
static __device__ inline void mlp_layer(_Float16 (*Y)[160], const float* w,
                                        const float* bias, int o0, int L0,
                                        int o1, int L1, int dstcol, bool relu,
                                        int lane) {
  const int Kdim = L0 + L1;
  const int nks  = (Kdim + 31) >> 5;
  v8f acc0 = vzero8(), acc1 = vzero8();
  for (int ks = 0; ks < nks; ++ks) {
    const v16h a = load_a_frag(&Y[0][0], 160, ks * 32, o0, L0, o1, L1, lane);
    acc0 = wmma_f16(a, load_b_frag_w(w, Kdim, 24, ks * 32, 0,  lane), acc0);
    acc1 = wmma_f16(a, load_b_frag_w(w, Kdim, 24, ks * 32, 16, lane), acc1);
  }
  const int n  = lane & 15;
  const int mb = (lane >> 4) * 8;
#pragma unroll
  for (int r = 0; r < 8; ++r) {
    float v = acc0[r] + bias[n];
    if (relu) v = fmaxf(v, 0.f);
    Y[mb + r][dstcol + n] = (_Float16)v;
  }
  if (n < 8) {
#pragma unroll
    for (int r = 0; r < 8; ++r) {
      float v = acc1[r] + bias[16 + n];
      if (relu) v = fmaxf(v, 0.f);
      Y[mb + r][dstcol + 16 + n] = (_Float16)v;
    }
  }
}

__global__ __launch_bounds__(256) void edgeconv_kernel(
    const float* __restrict__ feat, int fstride, int C,
    const int* __restrict__ knn,
    const float* __restrict__ w0, const float* __restrict__ b0,
    const float* __restrict__ w1, const float* __restrict__ b1,
    const float* __restrict__ w2, const float* __restrict__ b2,
    float* __restrict__ outbuf) {
  __shared__ _Float16 sy[8][16][160];
  __shared__ float sctr[8][48];
  const int lane = threadIdx.x & 31;
  const int wv   = threadIdx.x >> 5;
  const int p    = blockIdx.x * 8 + wv;
  const int b = p / NN, n = p % NN;
  const float* fb  = feat + (size_t)b * NN * fstride;
  const float* ctr = fb + (size_t)n * fstride;
  _Float16 (*Y)[160] = sy[wv];

  for (int t = lane; t < 16 * 160; t += 32) (&Y[0][0])[t] = (_Float16)0.f;
  for (int t = lane; t < C; t += 32) sctr[wv][t] = ctr[t];
  __syncthreads();

  const int* ki = knn + (size_t)(b * NN + n) * KNN_K;
  for (int j = 0; j < 16; ++j) {
    const int nb = ki[j];                       // wave-uniform scalar load
    const float* nf = fb + (size_t)nb * fstride;
    for (int t = lane; t < C; t += 32) {
      const float cv = sctr[wv][t];
      Y[j][t]     = (_Float16)cv;
      Y[j][C + t] = (_Float16)(nf[t] - cv);
    }
  }
  __syncthreads();

  // mlp0: K=2C -> h0 @ col 2C+24 (relu)
  mlp_layer(Y, w0, b0, 0, 2 * C, 0, 0, 2 * C + 24, true, lane);
  __syncthreads();
  // mlp1: [h0(24)@2C+24, ctr(C)@0] -> h1 @ col 2C (relu)
  mlp_layer(Y, w1, b1, 2 * C + 24, 24, 0, C, 2 * C, true, lane);
  __syncthreads();
  // mlp2: [h1,h0](48)@2C, ctr(C)@0 -> h2 @ col 2C-24 (no relu)
  mlp_layer(Y, w2, b2, 2 * C, 48, 0, C, 2 * C - 24, false, lane);
  __syncthreads();

  // max over 16 neighbors; logical layout [h2|h1|h0 (72), ctr (C)]
  const int Cout = 72 + C;
  const int base = 2 * C - 24;
  float* dst = outbuf + (size_t)(b * NN + n) * 480;
  for (int t = lane; t < Cout; t += 32) {
    const int col = (t < 72) ? (base + t) : (t - 72);
    float m = -3.0e38f;
#pragma unroll
    for (int j = 0; j < 16; ++j) m = fmaxf(m, (float)Y[j][col]);
    dst[t] = m;
  }
}

// ---------------------------------------------------------------------------
// prep conv: H = relu(X @ w + b), X:(B*N, xstride) first Cin cols, w:(Cin,48)
// one wave per 16 points, 3 WMMA N-tiles, K padded to 32-multiples.
// ---------------------------------------------------------------------------
__global__ __launch_bounds__(128) void prep_kernel(
    const float* __restrict__ X, int xstride, int Cin,
    const float* __restrict__ w, const float* __restrict__ bias,
    float* __restrict__ H) {
  __shared__ _Float16 sA[4][16][384];
  const int lane = threadIdx.x & 31;
  const int wv   = threadIdx.x >> 5;
  const int pt0  = (blockIdx.x * 4 + wv) * 16;
  _Float16 (*A)[384] = sA[wv];

  for (int t = lane; t < 16 * 384; t += 32) (&A[0][0])[t] = (_Float16)0.f;
  __syncthreads();
  for (int t = lane; t < 16 * Cin; t += 32) {
    const int r = t / Cin, c = t % Cin;
    A[r][c] = (_Float16)X[(size_t)(pt0 + r) * xstride + c];
  }
  __syncthreads();

  const int nks = (Cin + 31) >> 5;
  v8f acc[3] = {vzero8(), vzero8(), vzero8()};
  for (int ks = 0; ks < nks; ++ks) {
    const v16h a = load_a_frag(&A[0][0], 384, ks * 32, 0, Cin, 0, 0, lane);
#pragma unroll
    for (int nt = 0; nt < 3; ++nt)
      acc[nt] = wmma_f16(a, load_b_frag_w(w, Cin, 48, ks * 32, nt * 16, lane),
                         acc[nt]);
  }
  const int n  = lane & 15;
  const int mb = (lane >> 4) * 8;
#pragma unroll
  for (int nt = 0; nt < 3; ++nt) {
    const int col = nt * 16 + n;
#pragma unroll
    for (int r = 0; r < 8; ++r)
      H[(size_t)(pt0 + mb + r) * 48 + col] = fmaxf(acc[nt][r] + bias[col], 0.f);
  }
}

// ---------------------------------------------------------------------------
// transpose (B,N,480) -> (B,480,N)
// ---------------------------------------------------------------------------
__global__ void transpose_kernel(const float* __restrict__ Xall,
                                 float* __restrict__ out) {
  const size_t i = (size_t)blockIdx.x * blockDim.x + threadIdx.x;
  const size_t total = (size_t)BB * 480 * NN;
  if (i >= total) return;
  const int n = (int)(i % NN);
  const size_t t = i / NN;
  const int c = (int)(t % 480);
  const int b = (int)(t / 480);
  out[i] = Xall[((size_t)b * NN + n) * 480 + c];
}

// ---------------------------------------------------------------------------
extern "C" void kernel_launch(void* const* d_in, const int* in_sizes, int n_in,
                              void* d_out, int out_size, void* d_ws,
                              size_t ws_size, hipStream_t stream) {
  const float* xyz  = (const float*)d_in[0];
  const float* l0_w = (const float*)d_in[1];
  const float* l0_b = (const float*)d_in[2];
  const float* e1w0 = (const float*)d_in[3];  const float* e1b0 = (const float*)d_in[4];
  const float* e1w1 = (const float*)d_in[5];  const float* e1b1 = (const float*)d_in[6];
  const float* e1w2 = (const float*)d_in[7];  const float* e1b2 = (const float*)d_in[8];
  const float* p2w  = (const float*)d_in[9];  const float* p2b  = (const float*)d_in[10];
  const float* p3w  = (const float*)d_in[11]; const float* p3b  = (const float*)d_in[12];
  const float* p4w  = (const float*)d_in[13]; const float* p4b  = (const float*)d_in[14];
  const float* e2w0 = (const float*)d_in[15]; const float* e2b0 = (const float*)d_in[16];
  const float* e2w1 = (const float*)d_in[17]; const float* e2b1 = (const float*)d_in[18];
  const float* e2w2 = (const float*)d_in[19]; const float* e2b2 = (const float*)d_in[20];
  const float* e3w0 = (const float*)d_in[21]; const float* e3b0 = (const float*)d_in[22];
  const float* e3w1 = (const float*)d_in[23]; const float* e3b1 = (const float*)d_in[24];
  const float* e3w2 = (const float*)d_in[25]; const float* e3b2 = (const float*)d_in[26];
  const float* e4w0 = (const float*)d_in[27]; const float* e4b0 = (const float*)d_in[28];
  const float* e4w1 = (const float*)d_in[29]; const float* e4b1 = (const float*)d_in[30];
  const float* e4w2 = (const float*)d_in[31]; const float* e4b2 = (const float*)d_in[32];

  const size_t BN = (size_t)BB * NN;
  // workspace layout
  float* Xall = (float*)d_ws;                 // (B*N, 480) f32
  float* H    = Xall + BN * 480;              // (B*N, 48)  f32
  int*   idx  = (int*)(H + BN * 48);          // (B*N, 16)  i32
  const size_t need = (BN * 480 + BN * 48) * sizeof(float) + BN * 16 * sizeof(int);
  if (ws_size < need) return;

  const int knnGrid = BB * (NN / 16);         // 1 wave per 16-query tile
  const int ecGrid  = (int)(BN / 8);          // 8 points per 256-thread block
  const int prGrid  = (int)(BN / 64);         // 4x16 points per 128-thread block

  // layer0 -> Xall[:,456:480]
  layer0_kernel<<<(int)((BN + 255) / 256), 256, 0, stream>>>(xyz, l0_w, l0_b, Xall);

  // edge conv 1 (C=24) -> y1 @ Xall[:,360:456]
  knn_kernel<<<knnGrid, 32, 0, stream>>>(Xall + 456, 480, 24, idx);
  edgeconv_kernel<<<ecGrid, 256, 0, stream>>>(Xall + 456, 480, 24, idx,
                                              e1w0, e1b0, e1w1, e1b1, e1w2, e1b2,
                                              Xall + 360);
  // prep2: x1 = Xall[:,360:480] (120) -> H(48); edge conv 2 -> Xall[:,240:360]
  prep_kernel<<<prGrid, 128, 0, stream>>>(Xall + 360, 480, 120, p2w, p2b, H);
  knn_kernel<<<knnGrid, 32, 0, stream>>>(H, 48, 48, idx);
  edgeconv_kernel<<<ecGrid, 256, 0, stream>>>(H, 48, 48, idx,
                                              e2w0, e2b0, e2w1, e2b1, e2w2, e2b2,
                                              Xall + 240);
  // prep3: x2 = Xall[:,240:480] (240) -> H; edge conv 3 -> Xall[:,120:240]
  prep_kernel<<<prGrid, 128, 0, stream>>>(Xall + 240, 480, 240, p3w, p3b, H);
  knn_kernel<<<knnGrid, 32, 0, stream>>>(H, 48, 48, idx);
  edgeconv_kernel<<<ecGrid, 256, 0, stream>>>(H, 48, 48, idx,
                                              e3w0, e3b0, e3w1, e3b1, e3w2, e3b2,
                                              Xall + 120);
  // prep4: x3 = Xall[:,120:480] (360) -> H; edge conv 4 -> Xall[:,0:120]
  prep_kernel<<<prGrid, 128, 0, stream>>>(Xall + 120, 480, 360, p4w, p4b, H);
  knn_kernel<<<knnGrid, 32, 0, stream>>>(H, 48, 48, idx);
  edgeconv_kernel<<<ecGrid, 256, 0, stream>>>(H, 48, 48, idx,
                                              e4w0, e4b0, e4w1, e4b1, e4w2, e4b2,
                                              Xall + 0);

  // (B,N,480) -> (B,480,N)
  const size_t tot = BN * 480;
  transpose_kernel<<<(int)((tot + 255) / 256), 256, 0, stream>>>(Xall, (float*)d_out);
}
